// SSI_DDI_21964462752168
// MI455X (gfx1250) — matrix operand order repaired and checked by reading, not compiled, not run on hardware
//
#include <hip/hip_runtime.h>
#include <math.h>

// ---------------- problem constants ----------------
constexpr int N_NODES = 262144;           // B*NPG = 8192*32
constexpr int NEDGE   = 1048576;          // 4*N
constexpr int NGRAPH  = 8192;
constexpr int F_IN    = 70;
constexpr int D       = 128;
constexpr int NB      = 4;
constexpr int DK      = 64;
constexpr int K0PAD   = 96;               // F_IN=70 padded to multiple of 32
constexpr int EL      = NEDGE + N_NODES;  // edges + self loops

typedef unsigned short u16;
typedef unsigned int   u32;
typedef __attribute__((ext_vector_type(16))) __bf16 v16bf;
typedef __attribute__((ext_vector_type(8)))  float  v8f;

union Frag { uint4 q[2]; v16bf v; };
static_assert(sizeof(Frag) == 32, "frag size");

static __device__ __forceinline__ u16 f2bf(float f) {
  u32 u = __float_as_uint(f);
  u32 r = (u + 0x7FFFu + ((u >> 16) & 1u)) >> 16;
  return (u16)r;
}
static __device__ __forceinline__ float dec_ord(u32 u) {
  u = (u & 0x80000000u) ? (u & 0x7FFFFFFFu) : ~u;
  return __uint_as_float(u);
}
static __device__ __forceinline__ u32 enc_ord(float f) {
  u32 u = __float_as_uint(f);
  return ((int)u < 0) ? ~u : (u | 0x80000000u);
}

// ---------------- weight prep: bf16, transposed [n][k], padded K ----------------
__global__ __launch_bounds__(256) void prep_weights_kernel(
    const float* __restrict__ W0, const float* __restrict__ Wr, u16* __restrict__ wt) {
  int idx = blockIdx.x * 256 + threadIdx.x;
  constexpr int SZ0 = D * K0PAD;                 // 12288
  constexpr int TOT = SZ0 + (NB - 1) * D * D;    // 61440
  if (idx >= TOT) return;
  if (idx < SZ0) {
    int n = idx / K0PAD, k = idx % K0PAD;
    float v = (k < F_IN) ? W0[k * D + n] : 0.f;
    wt[idx] = f2bf(v);
  } else {
    int r = idx - SZ0;
    int layer = r / (D * D); int rr = r % (D * D);
    int n = rr / D, k = rr % D;
    wt[SZ0 + layer * D * D + n * D + k] = f2bf(Wr[layer * D * D + k * D + n]);
  }
}

// ---------------- input LayerNorm over F_IN=70, emit bf16 padded to 96 ----------------
__global__ __launch_bounds__(256) void lnorm_in_kernel(
    const float* __restrict__ x, const float* __restrict__ w, const float* __restrict__ b,
    u16* __restrict__ xb) {
  int gid = blockIdx.x * 256 + threadIdx.x;
  int node = gid >> 5, ln = gid & 31;
  const float* xr = x + (size_t)node * F_IN;
  float v0 = xr[ln], v1 = xr[ln + 32];
  float v2 = (ln < F_IN - 64) ? xr[ln + 64] : 0.f;
  float s = v0 + v1 + v2, q = v0 * v0 + v1 * v1 + v2 * v2;
  for (int o = 16; o > 0; o >>= 1) { s += __shfl_xor(s, o); q += __shfl_xor(q, o); }
  float mu = s * (1.f / F_IN);
  float var = q * (1.f / F_IN) - mu * mu;
  float rs = rsqrtf(var + 1e-5f);
  u16* xo = xb + (size_t)node * K0PAD;
  xo[ln]      = f2bf((v0 - mu) * rs * w[ln] + b[ln]);
  xo[ln + 32] = f2bf((v1 - mu) * rs * w[ln + 32] + b[ln + 32]);
  if (ln < F_IN - 64) xo[ln + 64] = f2bf((v2 - mu) * rs * w[ln + 64] + b[ln + 64]);
  else                xo[ln + 64] = 0;
}

// ---------------- WMMA GEMM: h[N,128] = xb[N,Kpad](bf16) @ Wt[Kpad,128](bf16) -----------
// Weight pre-transposed as wt[n][k]; staged in LDS, shared by all 8 waves.
__global__ __launch_bounds__(256) void gemm_kernel(
    const u16* __restrict__ xb, const u16* __restrict__ wt,
    float* __restrict__ h, int Kpad) {
  __shared__ u16 ldsW[D * D];  // up to 32 KB
  int tid = threadIdx.x;
  int nW4 = (D * Kpad) >> 3;   // uint4 count
  const uint4* wsrc = (const uint4*)wt;
  uint4* wdst = (uint4*)ldsW;
  for (int o = tid; o < nW4; o += 256) wdst[o] = wsrc[o];
  __syncthreads();

  int wave = tid >> 5, lane = tid & 31;
  int m15 = lane & 15, khalf = lane >> 4;
  int m = blockIdx.x * 128 + wave * 16 + m15;
  v8f acc[8];
  #pragma unroll
  for (int i = 0; i < 8; i++) acc[i] = (v8f){};

  const u16* arow = xb + (size_t)m * Kpad;
  int nsteps = Kpad >> 5;
  for (int ks = 0; ks < nsteps; ks++) {
    int k0 = ks * 32;
    Frag fa;
    fa.q[0] = *(const uint4*)(arow + k0 + khalf * 16);
    fa.q[1] = *(const uint4*)(arow + k0 + khalf * 16 + 8);
    #pragma unroll
    for (int nt = 0; nt < 8; nt++) {
      const u16* bp = ldsW + (nt * 16 + m15) * Kpad + k0 + khalf * 16;
      Frag fb;
      fb.q[0] = *(const uint4*)(bp);
      fb.q[1] = *(const uint4*)(bp + 8);
      acc[nt] = __builtin_amdgcn_wmma_f32_16x16x32_bf16(
          false, fa.v, false, fb.v, (short)0, acc[nt], false, false);
    }
  }
  int rbase = blockIdx.x * 128 + wave * 16 + khalf * 8;
  #pragma unroll
  for (int nt = 0; nt < 8; nt++)
    #pragma unroll
    for (int r = 0; r < 8; r++)
      h[(size_t)(rbase + r) * D + nt * 16 + m15] = acc[nt][r];
}

// ---------------- per-node attention coefficients (wave per node) ----------------
__global__ __launch_bounds__(256) void attcoef_kernel(
    const float* __restrict__ h, const float* __restrict__ as, const float* __restrict__ ad,
    float* __restrict__ a_src, float* __restrict__ a_dst) {
  int gid = blockIdx.x * 256 + threadIdx.x;
  int node = gid >> 5, ln = gid & 31;
  int f0 = ln * 4;
  float4 hv = *(const float4*)(h + (size_t)node * D + f0);
  float4 av = *(const float4*)(as + f0);
  float4 dv = *(const float4*)(ad + f0);
  float s = hv.x * av.x + hv.y * av.y + hv.z * av.z + hv.w * av.w;
  float d = hv.x * dv.x + hv.y * dv.y + hv.z * dv.z + hv.w * dv.w;
  for (int o = 8; o > 0; o >>= 1) { s += __shfl_xor(s, o); d += __shfl_xor(d, o); }
  if ((ln & 15) == 0) {
    int hd = ln >> 4;
    a_src[node * 2 + hd] = s;
    a_dst[node * 2 + hd] = d;
  }
}

// ---------------- edge softmax pass 1: segment max (ordered-uint atomicMax) -------------
__global__ __launch_bounds__(256) void edge_max_kernel(
    const int* __restrict__ src, const int* __restrict__ dst,
    const float* __restrict__ as, const float* __restrict__ ad, u32* __restrict__ m) {
  int idx = blockIdx.x * 256 + threadIdx.x;
  if (idx >= EL) return;
  int s, d;
  if (idx < NEDGE) { s = src[idx]; d = dst[idx]; } else { s = d = idx - NEDGE; }
  #pragma unroll
  for (int hd = 0; hd < 2; hd++) {
    float a = as[s * 2 + hd] + ad[d * 2 + hd];
    a = (a >= 0.f) ? a : 0.2f * a;
    atomicMax(m + d * 2 + hd, enc_ord(a));
  }
}

// ---------------- edge softmax pass 2: exp + segment sum ----------------
__global__ __launch_bounds__(256) void edge_exp_kernel(
    const int* __restrict__ src, const int* __restrict__ dst,
    const float* __restrict__ as, const float* __restrict__ ad,
    const u32* __restrict__ m, float* __restrict__ ex, float* __restrict__ den) {
  int idx = blockIdx.x * 256 + threadIdx.x;
  if (idx >= EL) return;
  int s, d;
  if (idx < NEDGE) { s = src[idx]; d = dst[idx]; } else { s = d = idx - NEDGE; }
  #pragma unroll
  for (int hd = 0; hd < 2; hd++) {
    float a = as[s * 2 + hd] + ad[d * 2 + hd];
    a = (a >= 0.f) ? a : 0.2f * a;
    float e = __expf(a - dec_ord(m[d * 2 + hd]));
    ex[(size_t)idx * 2 + hd] = e;
    atomicAdd(den + d * 2 + hd, e);
  }
}

// ---------------- init out to bias (replaces zero + bias add) ----------------
__global__ __launch_bounds__(256) void fill_bias_kernel(
    float* __restrict__ outb, const float* __restrict__ gb) {
  int idx = blockIdx.x * 256 + threadIdx.x;
  outb[idx] = gb[idx & (D - 1)];
}

// ---------------- weighted message aggregation (32 lanes per edge) ----------------
__global__ __launch_bounds__(256) void edge_aggr_kernel(
    const int* __restrict__ src, const int* __restrict__ dst,
    const float* __restrict__ h, const float* __restrict__ ex,
    const float* __restrict__ den, float* __restrict__ outb) {
  int gid = blockIdx.x * 256 + threadIdx.x;
  int idx = gid >> 5, ln = gid & 31;
  if (idx >= EL) return;
  int s, d;
  if (idx < NEDGE) { s = src[idx]; d = dst[idx]; } else { s = d = idx - NEDGE; }
  int f0 = ln * 4, hd = ln >> 4;
  float w = ex[(size_t)idx * 2 + hd] / (den[d * 2 + hd] + 1e-16f);
  float4 hv = *(const float4*)(h + (size_t)s * D + f0);
  float* ob = outb + (size_t)d * D + f0;
  atomicAdd(ob + 0, hv.x * w);
  atomicAdd(ob + 1, hv.y * w);
  atomicAdd(ob + 2, hv.z * w);
  atomicAdd(ob + 3, hv.w * w);
}

// ---------------- readout neighbor sum (no self loops) ----------------
__global__ __launch_bounds__(256) void raggr_kernel(
    const int* __restrict__ src, const int* __restrict__ dst,
    const float* __restrict__ xp, float* __restrict__ aggr) {
  int gid = blockIdx.x * 256 + threadIdx.x;
  int idx = gid >> 5, ln = gid & 31;
  if (idx >= NEDGE) return;
  int s = src[idx], d = dst[idx];
  int f0 = ln * 4;
  float4 v = *(const float4*)(xp + (size_t)s * D + f0);
  float* ob = aggr + (size_t)d * D + f0;
  atomicAdd(ob + 0, v.x);
  atomicAdd(ob + 1, v.y);
  atomicAdd(ob + 2, v.z);
  atomicAdd(ob + 3, v.w);
}

// ---------------- SAG score (wave per node) ----------------
__global__ __launch_bounds__(256) void score_kernel(
    const float* __restrict__ aggr, const float* __restrict__ xp,
    const float* __restrict__ rw, const float* __restrict__ rb,
    const float* __restrict__ rootw, float* __restrict__ score) {
  int gid = blockIdx.x * 256 + threadIdx.x;
  int node = gid >> 5, ln = gid & 31;
  const float* ar = aggr + (size_t)node * D;
  const float* xr = xp + (size_t)node * D;
  float acc = 0.f;
  for (int f = ln; f < D; f += 32) acc += ar[f] * rw[f] + xr[f] * rootw[f];
  for (int o = 16; o > 0; o >>= 1) acc += __shfl_xor(acc, o);
  if (ln == 0) score[node] = acc + rb[0];
}

// ---------------- per-graph softmax pool (graphs are contiguous 32-node blocks) ---------
__global__ __launch_bounds__(128) void pool_kernel(
    const float* __restrict__ score, const float* __restrict__ xp,
    float* __restrict__ repr, int layer) {
  __shared__ float ssc[32];
  int b = blockIdx.x, tid = threadIdx.x;
  int nodeBase = b * 32;
  if (tid < 32) {
    float sc = score[nodeBase + tid];
    float mx = sc;
    for (int o = 16; o > 0; o >>= 1) mx = fmaxf(mx, __shfl_xor(mx, o));
    float e = __expf(sc - mx);
    float sm = e;
    for (int o = 16; o > 0; o >>= 1) sm += __shfl_xor(sm, o);
    ssc[tid] = e / (sm + 1e-16f);
  }
  __syncthreads();
  float acc = 0.f;
  for (int j = 0; j < 32; j++) acc += xp[(size_t)(nodeBase + j) * D + tid] * ssc[j];
  repr[(size_t)b * (NB * D) + layer * D + tid] = acc;
}

// ---------------- inter-layer LayerNorm + ELU, emit bf16 for next GEMM -----------------
__global__ __launch_bounds__(256) void lnorm_elu_kernel(
    const float* __restrict__ xp, const float* __restrict__ w, const float* __restrict__ b,
    u16* __restrict__ xb) {
  int gid = blockIdx.x * 256 + threadIdx.x;
  int node = gid >> 5, ln = gid & 31;
  int f0 = ln * 4;
  float4 v = *(const float4*)(xp + (size_t)node * D + f0);
  float s = v.x + v.y + v.z + v.w;
  float q = v.x * v.x + v.y * v.y + v.z * v.z + v.w * v.w;
  for (int o = 16; o > 0; o >>= 1) { s += __shfl_xor(s, o); q += __shfl_xor(q, o); }
  float mu = s * (1.f / D);
  float var = q * (1.f / D) - mu * mu;
  float rs = rsqrtf(var + 1e-5f);
  float4 wv = *(const float4*)(w + f0);
  float4 bv = *(const float4*)(b + f0);
  float y0 = (v.x - mu) * rs * wv.x + bv.x;
  float y1 = (v.y - mu) * rs * wv.y + bv.y;
  float y2 = (v.z - mu) * rs * wv.z + bv.z;
  float y3 = (v.w - mu) * rs * wv.w + bv.w;
  y0 = (y0 > 0.f) ? y0 : __expf(y0) - 1.f;
  y1 = (y1 > 0.f) ? y1 : __expf(y1) - 1.f;
  y2 = (y2 > 0.f) ? y2 : __expf(y2) - 1.f;
  y3 = (y3 > 0.f) ? y3 : __expf(y3) - 1.f;
  uint2 packed;
  packed.x = (u32)f2bf(y0) | ((u32)f2bf(y1) << 16);
  packed.y = (u32)f2bf(y2) | ((u32)f2bf(y3) << 16);
  *(uint2*)(xb + (size_t)node * D + f0) = packed;
}

// ---------------- co-attention + RESCAL per graph ----------------
__global__ __launch_bounds__(128) void final_kernel(
    const float* __restrict__ rh, const float* __restrict__ rt,
    const int* __restrict__ rels, const float* __restrict__ rel_emb,
    const float* __restrict__ wq, const float* __restrict__ wk,
    const float* __restrict__ cob, const float* __restrict__ coa,
    float* __restrict__ out) {
  __shared__ float sh[NB][D], st[NB][D], tmp[NB][D];
  __shared__ float ky[NB][DK], qu[NB][DK];
  __shared__ float satt[NB * NB];
  __shared__ float red[128];
  int b = blockIdx.x, tid = threadIdx.x;
  const float* rhb = rh + (size_t)b * NB * D;
  const float* rtb = rt + (size_t)b * NB * D;
  for (int o = tid; o < NB * D; o += 128) {
    sh[o >> 7][o & (D - 1)] = rhb[o];
    st[o >> 7][o & (D - 1)] = rtb[o];
  }
  __syncthreads();
  // keys = rh@wk, queries = rt@wq  (raw reprs)
  for (int o = tid; o < NB * DK; o += 128) {
    int i = o >> 6, qq = o & (DK - 1);
    float accK = 0.f, accQ = 0.f;
    for (int d = 0; d < D; d++) {
      accK += sh[i][d] * wk[d * DK + qq];
      accQ += st[i][d] * wq[d * DK + qq];
    }
    ky[i][qq] = accK; qu[i][qq] = accQ;
  }
  __syncthreads();
  if (tid < NB * NB) {
    int i = tid >> 2, j = tid & 3;
    float a = 0.f;
    for (int qq = 0; qq < DK; qq++) a += tanhf(ky[i][qq] + qu[j][qq] + cob[qq]) * coa[qq];
    satt[tid] = a;
  }
  // row l2-normalize sh and st (warp per row)
  {
    int wv = tid >> 5, ln = tid & 31;
    float s1 = 0.f, s2 = 0.f;
    for (int d = ln; d < D; d += 32) {
      float a = sh[wv][d]; s1 += a * a;
      float c = st[wv][d]; s2 += c * c;
    }
    for (int o = 16; o > 0; o >>= 1) { s1 += __shfl_xor(s1, o); s2 += __shfl_xor(s2, o); }
    float i1 = 1.f / fmaxf(sqrtf(s1), 1e-12f);
    float i2 = 1.f / fmaxf(sqrtf(s2), 1e-12f);
    for (int d = ln; d < D; d += 32) { sh[wv][d] *= i1; st[wv][d] *= i2; }
  }
  __syncthreads();
  // R pass: tmp[j][d] = sum_e R[d][e]*t[j][e], plus global norm of R
  const float* R = rel_emb + (size_t)rels[b] * (D * D);
  {
    const float* row = R + tid * D;
    float a0 = 0.f, a1 = 0.f, a2 = 0.f, a3 = 0.f, nsq = 0.f;
    for (int e = 0; e < D; e++) {
      float r = row[e];
      nsq += r * r;
      a0 += r * st[0][e]; a1 += r * st[1][e]; a2 += r * st[2][e]; a3 += r * st[3][e];
    }
    tmp[0][tid] = a0; tmp[1][tid] = a1; tmp[2][tid] = a2; tmp[3][tid] = a3;
    red[tid] = nsq;
  }
  for (int s = 64; s > 0; s >>= 1) { __syncthreads(); if (tid < s) red[tid] += red[tid + s]; }
  __syncthreads();
  float invn = 1.f / fmaxf(sqrtf(red[0]), 1e-12f);
  __syncthreads();
  // scores[i][j] = invn * h[i] . tmp[j]; result = sum att*scores
  float part = 0.f;
  {
    int pair = tid >> 3, l8 = tid & 7;
    int i = pair >> 2, j = pair & 3;
    float sc = 0.f;
    for (int d = l8; d < D; d += 8) sc += sh[i][d] * tmp[j][d];
    for (int o = 4; o > 0; o >>= 1) sc += __shfl_xor(sc, o);
    if (l8 == 0) part = satt[pair] * sc * invn;
  }
  red[tid] = part;
  for (int s = 64; s > 0; s >>= 1) { __syncthreads(); if (tid < s) red[tid] += red[tid + s]; }
  __syncthreads();
  if (tid == 0) out[b] = red[0];
}

// ---------------- workspace layout ----------------
constexpr size_t AL(size_t x) { return (x + 255) & ~size_t(255); }
constexpr size_t SZ_XB   = AL((size_t)N_NODES * D * 2);
constexpr size_t SZ_F    = AL((size_t)N_NODES * D * 4);
constexpr size_t SZ_A    = AL((size_t)N_NODES * 2 * 4);
constexpr size_t SZ_EX   = AL((size_t)EL * 2 * 4);
constexpr size_t SZ_SC   = AL((size_t)N_NODES * 4);
constexpr size_t SZ_R    = AL((size_t)NGRAPH * NB * D * 4);
constexpr size_t OFF_XB   = 0;
constexpr size_t OFF_H    = OFF_XB + SZ_XB;
constexpr size_t OFF_XP   = OFF_H + SZ_F;
constexpr size_t OFF_AGGR = OFF_XP + SZ_F;
constexpr size_t OFF_ASRC = OFF_AGGR + SZ_F;
constexpr size_t OFF_ADST = OFF_ASRC + SZ_A;
constexpr size_t OFF_M    = OFF_ADST + SZ_A;
constexpr size_t OFF_DEN  = OFF_M + (size_t)N_NODES * 2 * 4;  // adjacent to M (one memset)
constexpr size_t OFF_EX   = AL(OFF_DEN + (size_t)N_NODES * 2 * 4);
constexpr size_t OFF_SCR  = OFF_EX + SZ_EX;
constexpr size_t OFF_RH   = OFF_SCR + SZ_SC;
constexpr size_t OFF_RT   = OFF_RH + SZ_R;
constexpr size_t OFF_WT   = OFF_RT + SZ_R;

extern "C" void kernel_launch(void* const* d_in, const int* in_sizes, int n_in,
                              void* d_out, int out_size, void* d_ws, size_t ws_size,
                              hipStream_t stream) {
  (void)in_sizes; (void)n_in; (void)out_size; (void)ws_size;
  const float* xh       = (const float*)d_in[0];
  const float* xt       = (const float*)d_in[1];
  const int*   eh       = (const int*)d_in[2];
  const int*   et       = (const int*)d_in[3];
  const int*   rels     = (const int*)d_in[5];
  const float* W0       = (const float*)d_in[6];
  const float* Wr       = (const float*)d_in[7];
  const float* att_src  = (const float*)d_in[8];
  const float* att_dst  = (const float*)d_in[9];
  const float* gat_b    = (const float*)d_in[10];
  const float* sag_rw   = (const float*)d_in[11];
  const float* sag_rb   = (const float*)d_in[12];
  const float* sag_root = (const float*)d_in[13];
  const float* nw       = (const float*)d_in[14];
  const float* nb2      = (const float*)d_in[15];
  const float* inw      = (const float*)d_in[16];
  const float* inb      = (const float*)d_in[17];
  const float* wq       = (const float*)d_in[18];
  const float* wk       = (const float*)d_in[19];
  const float* cob      = (const float*)d_in[20];
  const float* coa      = (const float*)d_in[21];
  const float* rel_emb  = (const float*)d_in[22];

  char* ws = (char*)d_ws;
  u16*   xbp   = (u16*)(ws + OFF_XB);
  float* hp    = (float*)(ws + OFF_H);
  float* xpp   = (float*)(ws + OFF_XP);
  float* aggrp = (float*)(ws + OFF_AGGR);
  float* asrcp = (float*)(ws + OFF_ASRC);
  float* adstp = (float*)(ws + OFF_ADST);
  u32*   mp    = (u32*)(ws + OFF_M);
  float* denp  = (float*)(ws + OFF_DEN);
  float* exp_  = (float*)(ws + OFF_EX);
  float* scp   = (float*)(ws + OFF_SCR);
  u16*   wtp   = (u16*)(ws + OFF_WT);

  prep_weights_kernel<<<240, 256, 0, stream>>>(W0, Wr, wtp);

  float* repr[2] = { (float*)(ws + OFF_RH), (float*)(ws + OFF_RT) };
  const float* xin[2] = { xh, xt };
  const int* edges[2] = { eh, et };

  for (int side = 0; side < 2; side++) {
    const int* src = edges[side];
    const int* dst = edges[side] + NEDGE;
    lnorm_in_kernel<<<N_NODES / 8, 256, 0, stream>>>(xin[side], inw, inb, xbp);
    for (int i = 0; i < NB; i++) {
      int Kpad = (i == 0) ? K0PAD : D;
      const u16* wti = wtp + ((i == 0) ? 0 : (D * K0PAD + (i - 1) * D * D));
      gemm_kernel<<<N_NODES / 128, 256, 0, stream>>>(xbp, wti, hp, Kpad);
      attcoef_kernel<<<N_NODES / 8, 256, 0, stream>>>(hp, att_src + i * D, att_dst + i * D,
                                                      asrcp, adstp);
      hipMemsetAsync(ws + OFF_M, 0, (size_t)N_NODES * 2 * 4 * 2, stream);  // m(-inf enc)+den
      edge_max_kernel<<<EL / 256, 256, 0, stream>>>(src, dst, asrcp, adstp, mp);
      edge_exp_kernel<<<EL / 256, 256, 0, stream>>>(src, dst, asrcp, adstp, mp, exp_, denp);
      fill_bias_kernel<<<N_NODES * D / 256, 256, 0, stream>>>(xpp, gat_b + i * D);
      edge_aggr_kernel<<<EL / 8, 256, 0, stream>>>(src, dst, hp, exp_, denp, xpp);
      // readout
      hipMemsetAsync(ws + OFF_AGGR, 0, (size_t)N_NODES * D * 4, stream);
      raggr_kernel<<<NEDGE / 8, 256, 0, stream>>>(src, dst, xpp, aggrp);
      score_kernel<<<N_NODES / 8, 256, 0, stream>>>(aggrp, xpp, sag_rw + i * D, sag_rb + i,
                                                    sag_root + i * D, scp);
      pool_kernel<<<NGRAPH, 128, 0, stream>>>(scp, xpp, repr[side], i);
      if (i < NB - 1)
        lnorm_elu_kernel<<<N_NODES / 8, 256, 0, stream>>>(xpp, nw + i * D, nb2 + i * D, xbp);
    }
  }
  final_kernel<<<NGRAPH, 128, 0, stream>>>(repr[0], repr[1], rels, rel_emb,
                                           wq, wk, cob, coa, (float*)d_out);
}
